// _ExtendedGraphConvKerasModel_82136954569142
// MI455X (gfx1250) — compile-verified, matrix operand order checked by reading, not run on hardware
//
#include <hip/hip_runtime.h>
#include <math.h>

#define MAX_DEG 10
#define N_ATOMS 1000000
#define BATCH   32768
#define EPS_BN  1e-3f

static const int H_COUNTS[11] = {10000,250000,300000,250000,120000,40000,15000,8000,4000,2000,1000};
static const int H_STARTS[11] = {0,10000,260000,560000,810000,930000,970000,985000,993000,997000,999000};

typedef __attribute__((ext_vector_type(16))) __bf16       v16bf;
typedef __attribute__((ext_vector_type(8)))  float        v8f;
typedef __attribute__((ext_vector_type(4)))  unsigned int u32x4;

union BFRAG { v16bf v; u32x4 q[2]; };
union F32x8 { v8f   v; float f[8]; };

__device__ __forceinline__ unsigned short f2bf(float f) {
    union { float f; unsigned int u; } x; x.f = f;
    unsigned int r = x.u + 0x7FFFu + ((x.u >> 16) & 1u);   // RNE
    return (unsigned short)(r >> 16);
}
__device__ __forceinline__ float bflo(unsigned int p) { return __uint_as_float(p << 16); }
__device__ __forceinline__ float bfhi(unsigned int p) { return __uint_as_float(p & 0xFFFF0000u); }
__device__ __forceinline__ unsigned int bfpack(float a, float b) {
    return (unsigned int)f2bf(a) | ((unsigned int)f2bf(b) << 16);
}

// ---------------------------------------------------------------------------
// WMMA GEMM, bf16 A (row-major, stride K) x pre-packed bf16 B, f32 accumulate.
// Packed B layout: [ntile][ktile][lane][16 bf16]  (ntile count == blockDim/32).
// One wave = one 16x16 tile. K must be a multiple of 32. Unconditional b128
// loads (rows clamped); only stores/bias are guarded. HAS_A2 selects the
// dual-matrix (self + neighbor-sum) conv path at compile time.
// ---------------------------------------------------------------------------
template <bool HAS_A2>
__global__ void gemm_bf16_kernel(const unsigned short* __restrict__ A,
                                 const unsigned short* __restrict__ A2,
                                 const unsigned short* __restrict__ Wp,
                                 const unsigned short* __restrict__ Wp2,
                                 const float* __restrict__ b1, const float* __restrict__ b2,
                                 const float* __restrict__ bng, const float* __restrict__ bnb,
                                 const float* __restrict__ bnm, const float* __restrict__ bnv,
                                 unsigned short* __restrict__ Cb, float* __restrict__ Cf,
                                 int rows, int K, int N, int doRelu)
{
    const int lane = threadIdx.x & 31;
    const int wave = threadIdx.x >> 5;
    const int h    = lane >> 4;
    const int l15  = lane & 15;
    const int rowBase = blockIdx.x * 16;
    const int KT = K >> 5;

    int arow = rowBase + l15;
    if (arow > rows - 1) arow = rows - 1;                 // clamp -> branchless loads

    const unsigned short* aP  = A + (size_t)arow * K + 8 * h;
    const unsigned short* bP  = Wp + ((size_t)(wave * KT) * 32 + lane) * 16;
    const unsigned short* aP2 = HAS_A2 ? A2 + (size_t)arow * K + 8 * h : (const unsigned short*)0;
    const unsigned short* bP2 = HAS_A2 ? Wp2 + ((size_t)(wave * KT) * 32 + lane) * 16
                                       : (const unsigned short*)0;

    F32x8 acc;
#pragma unroll
    for (int j = 0; j < 8; ++j) acc.f[j] = 0.0f;

    for (int kt = 0; kt < KT; ++kt) {
        BFRAG a, b;
        a.q[0] = *(const u32x4*)(aP);
        a.q[1] = *(const u32x4*)(aP + 16);
        b.q[0] = *(const u32x4*)(bP);
        b.q[1] = *(const u32x4*)(bP + 8);
        acc.v = __builtin_amdgcn_wmma_f32_16x16x32_bf16(false, a.v, false, b.v,
                                                        (short)0, acc.v, false, false);
        if (HAS_A2) {
            BFRAG a2, b2f;
            a2.q[0]  = *(const u32x4*)(aP2);
            a2.q[1]  = *(const u32x4*)(aP2 + 16);
            b2f.q[0] = *(const u32x4*)(bP2);
            b2f.q[1] = *(const u32x4*)(bP2 + 8);
            acc.v = __builtin_amdgcn_wmma_f32_16x16x32_bf16(false, a2.v, false, b2f.v,
                                                            (short)0, acc.v, false, false);
            aP2 += 32; bP2 += 512;
        }
        aP += 32; bP += 512;
    }

    const int c = wave * 16 + l15;                        // D: col = lane&15, row = 8h+j
    if (c < N) {
        float bias = b1 ? b1[c] : 0.0f;
        if (b2) bias += b2[c];
        float g = 1.0f, bb = 0.0f, mu = 0.0f, iv = 1.0f;
        if (bng) { g = bng[c]; bb = bnb[c]; mu = bnm[c]; iv = rsqrtf(bnv[c] + EPS_BN); }
#pragma unroll
        for (int j = 0; j < 8; ++j) {
            int r = rowBase + 8 * h + j;
            if (r < rows) {
                float v = acc.f[j] + bias;
                if (doRelu) v = fmaxf(v, 0.0f);
                if (bng)    v = (v - mu) * iv * g + bb;
                if (Cb) Cb[(size_t)r * N + c] = f2bf(v);
                else    Cf[(size_t)r * N + c] = v;
            }
        }
    }
}

// Pack f32 weight matrices -> bf16 WMMA B-fragment layout, zero padded.
__global__ void pack_w_kernel(const float* __restrict__ W, unsigned short* __restrict__ out,
                              int nmat, int Ksrc, int Nsrc, int KT, int NT)
{
    int total = nmat * NT * KT * 512;
    int t = blockIdx.x * blockDim.x + threadIdx.x;
    if (t >= total) return;
    int mat = t / (NT * KT * 512); int r = t % (NT * KT * 512);
    int nt  = r / (KT * 512);          r %= (KT * 512);
    int kt  = r / 512;                 r %= 512;
    int lane = r >> 4, e = r & 15;
    int hh = lane >> 4, l15 = lane & 15;
    int k = kt * 32 + ((e < 8) ? (8 * hh + e) : (16 + 8 * hh + (e - 8)));
    int n = nt * 16 + l15;
    float v = (k < Ksrc && n < Nsrc) ? W[((size_t)mat * Ksrc + k) * Nsrc + n] : 0.0f;
    out[t] = f2bf(v);
}

// atoms f32 (1M x 75) -> bf16 padded (1M x 96), pair-packed writes
__global__ void pack_atoms_kernel(const float* __restrict__ a, unsigned short* __restrict__ xa,
                                  int rows)
{
    int t = blockIdx.x * blockDim.x + threadIdx.x;
    if (t >= rows * 48) return;
    int row = t / 48, u = t % 48, f = 2 * u;
    float v0 = (f     < 75) ? a[(size_t)row * 75 + f]     : 0.0f;
    float v1 = (f + 1 < 75) ? a[(size_t)row * 75 + f + 1] : 0.0f;
    ((unsigned int*)xa)[t] = bfpack(v0, v1);
}

// rel[row,:] = sum_j x[adj[row,j],:]  (bf16 in/out, one wave per row)
__global__ void gather_sum_kernel(const unsigned short* __restrict__ x,
                                  const int* __restrict__ adj,
                                  unsigned short* __restrict__ rel,
                                  int rows, int deg, int F2 /* = F/2 uints */)
{
    int wave = blockIdx.x * (blockDim.x >> 5) + (threadIdx.x >> 5);
    int lane = threadIdx.x & 31;
    if (wave >= rows) return;
    const unsigned int* xr = (const unsigned int*)x;
    const int* a = adj + (size_t)wave * deg;
    for (int u = lane; u < F2; u += 32) {
        float s0 = 0.0f, s1 = 0.0f;
        for (int j = 0; j < deg; ++j) {
            unsigned int p = xr[(size_t)a[j] * F2 + u];
            s0 += bflo(p); s1 += bfhi(p);
        }
        ((unsigned int*)rel)[(size_t)wave * F2 + u] = bfpack(s0, s1);
    }
}

// out[start+row,:] = max(x[start+row,:], max_j x[adj[row,j],:])  (bf16 pairs)
__global__ void pool_kernel(const unsigned short* __restrict__ x, const int* __restrict__ adj,
                            unsigned short* __restrict__ out, int rows, int deg, int start,
                            int F2)
{
    int t = blockIdx.x * blockDim.x + threadIdx.x;
    int row = t / F2, u = t % F2;
    if (row >= rows) return;
    const unsigned int* xr = (const unsigned int*)x;
    unsigned int p = xr[(size_t)(start + row) * F2 + u];
    float m0 = bflo(p), m1 = bfhi(p);
    const int* a = adj + (size_t)row * deg;
    for (int j = 0; j < deg; ++j) {
        unsigned int q = xr[(size_t)a[j] * F2 + u];
        m0 = fmaxf(m0, bflo(q)); m1 = fmaxf(m1, bfhi(q));
    }
    ((unsigned int*)out)[(size_t)(start + row) * F2 + u] = bfpack(m0, m1);
}

__global__ void seg_init_kernel(float* __restrict__ s, float* __restrict__ m, int n) {
    int t = blockIdx.x * blockDim.x + threadIdx.x;
    if (t < n) { s[t] = 0.0f; m[t] = -INFINITY; }
}

__device__ __forceinline__ void atomicMaxF(float* addr, float val) {
    int* ia = (int*)addr;
    int old = __float_as_int(*addr);
    while (__int_as_float(old) < val) {
        int assumed = old;
        old = atomicCAS(ia, assumed, __float_as_int(val));
        if (old == assumed) break;
    }
}

__global__ void segment_kernel(const float* __restrict__ x, const int* __restrict__ mem,
                               float* __restrict__ s, float* __restrict__ m, int atoms, int F)
{
    int t = blockIdx.x * blockDim.x + threadIdx.x;
    if (t >= atoms * F) return;
    int a = t / F, f = t % F;
    float v = x[(size_t)a * F + f];
    int g = mem[a];
    atomicAdd(&s[(size_t)g * F + f], v);
    atomicMaxF(&m[(size_t)g * F + f], v);
}

// nf = tanh(concat(sum, max)); write f32 (output) and bf16 (next-GEMM input)
__global__ void nf_kernel(const float* __restrict__ s, const float* __restrict__ m,
                          float* __restrict__ nf, unsigned short* __restrict__ nfb,
                          int nseg, int F)
{
    int t = blockIdx.x * blockDim.x + threadIdx.x;
    if (t >= nseg * 2 * F) return;
    int seg = t / (2 * F), c = t % (2 * F);
    float v = (c < F) ? s[(size_t)seg * F + c] : m[(size_t)seg * F + (c - F)];
    v = tanhf(v);
    nf[t] = v;
    nfb[t] = f2bf(v);
}

__global__ void softmax_kernel(const float* __restrict__ logits, float* __restrict__ out,
                               int npairs)
{
    int t = blockIdx.x * blockDim.x + threadIdx.x;
    if (t >= npairs) return;
    float l0 = logits[2 * t], l1 = logits[2 * t + 1];
    float mx = fmaxf(l0, l1);
    float e0 = expf(l0 - mx), e1 = expf(l1 - mx);
    float inv = 1.0f / (e0 + e1);
    out[2 * t] = e0 * inv; out[2 * t + 1] = e1 * inv;
}

// ---------------------------------------------------------------------------
extern "C" void kernel_launch(void* const* d_in, const int* in_sizes, int n_in,
                              void* d_out, int out_size, void* d_ws, size_t ws_size,
                              hipStream_t stream)
{
    typedef unsigned short u16;
    const float* atoms      = (const float*)d_in[0];
    const int*   membership = (const int*)d_in[2];
    const int*   adj[10];
    for (int i = 0; i < 10; ++i) adj[i] = (const int*)d_in[4 + i];
    const float* gcW[2] = {(const float*)d_in[14], (const float*)d_in[16]};
    const float* gcB[2] = {(const float*)d_in[15], (const float*)d_in[17]};
    const float* bnP[3][4];
    for (int j = 0; j < 3; ++j)
        for (int k = 0; k < 4; ++k) bnP[j][k] = (const float*)d_in[18 + 4 * j + k];
    const float* dense_W = (const float*)d_in[30];
    const float* dense_b = (const float*)d_in[31];
    const float* fd0_W   = (const float*)d_in[32];
    const float* fd0_b   = (const float*)d_in[33];
    const float* rd_W    = (const float*)d_in[34];
    const float* rd_b    = (const float*)d_in[35];

    char* ws = (char*)d_ws;
    u16*   XA    = (u16*)  (ws);                    // 1M x 96 bf16 (192 MB)
    u16*   REL   = (u16*)  (ws + 192000000ULL);     // 1M x 96 bf16 (192 MB)
    u16*   CONV  = (u16*)  (ws + 384000000ULL);     // 1M x 64 bf16 (128 MB)
    u16*   POOL  = (u16*)  (ws + 512000000ULL);     // 1M x 64 bf16 (128 MB)
    float* DENSE = (float*)(ws);                    // 1M x 128 f32 (512 MB) reuses XA+REL+CONV
    float* SEGS  = (float*)(ws + 640000000ULL);     // 32768 x 128
    float* SEGM  = (float*)(ws + 656777216ULL);     // 32768 x 128
    u16*   NFB   = (u16*)  (ws + 673554432ULL);     // 32768 x 256 bf16
    u16*   HB    = (u16*)  (ws + 690331648ULL);     // 32768 x 256 bf16
    u16*   WPK   = (u16*)  (ws + 707108864ULL);     // packed weights (~0.6 MB)

    u16* wp_gc0   = WPK;                  // 21 mats, KT=3, NT=4 -> 6144 u16/mat
    u16* wp_gc1   = WPK + 129024;         // 21 mats, KT=2, NT=4 -> 4096 u16/mat
    u16* wp_dense = WPK + 215040;         // KT=2,  NT=8  -> 8192 u16
    u16* wp_fd0   = WPK + 223232;         // KT=8,  NT=16 -> 65536 u16
    u16* wp_rd    = WPK + 288768;         // KT=8,  NT=2  -> 8192 u16

    float* outp       = (float*)d_out;
    float* OUT_SOFT   = outp;                        // 32768 x 24
    float* OUT_LOGITS = outp + (size_t)BATCH * 24;   // 32768 x 24
    float* NF         = outp + (size_t)BATCH * 48;   // 32768 x 256

    // -------- prepack weights + input activations (cheap, every call) --------
    {
        int t;
        t = 21 * 4 * 3 * 512;  pack_w_kernel<<<(t + 255) / 256, 256, 0, stream>>>(gcW[0], wp_gc0, 21, 75, 64, 3, 4);
        t = 21 * 4 * 2 * 512;  pack_w_kernel<<<(t + 255) / 256, 256, 0, stream>>>(gcW[1], wp_gc1, 21, 64, 64, 2, 4);
        t = 8 * 2 * 512;       pack_w_kernel<<<(t + 255) / 256, 256, 0, stream>>>(dense_W, wp_dense, 1, 64, 128, 2, 8);
        t = 16 * 8 * 512;      pack_w_kernel<<<(t + 255) / 256, 256, 0, stream>>>(fd0_W, wp_fd0, 1, 256, 256, 8, 16);
        t = 2 * 8 * 512;       pack_w_kernel<<<(t + 255) / 256, 256, 0, stream>>>(rd_W, wp_rd, 1, 256, 24, 8, 2);
        t = N_ATOMS * 48;      pack_atoms_kernel<<<(t + 255) / 256, 256, 0, stream>>>(atoms, XA, N_ATOMS);
    }

    // -------------------- two graph-conv + pool layers --------------------
    for (int layer = 0; layer < 2; ++layer) {
        const u16* xin  = (layer == 0) ? XA : POOL;
        const int  K    = (layer == 0) ? 96 : 64;     // padded K
        const int  F2   = K >> 1;
        const int  cout = 64;
        u16* wbase = (layer == 0) ? wp_gc0 : wp_gc1;
        const float* B = gcB[layer];
        const size_t matStride = (size_t)4 /*NT*/ * (K >> 5) * 512;

        for (int d = 1; d <= MAX_DEG; ++d) {
            int rows = H_COUNTS[d];
            gather_sum_kernel<<<(rows + 7) / 8, 256, 0, stream>>>(
                xin, adj[d - 1], REL + (size_t)H_STARTS[d] * K, rows, d, F2);
        }
        // deg-0 block: single-matrix GEMM
        {
            int rows = H_COUNTS[0];
            const u16*   Wself = wbase + (size_t)(2 * MAX_DEG) * matStride;
            const float* bself = B + (size_t)(2 * MAX_DEG) * cout;
            gemm_bf16_kernel<false><<<(rows + 15) / 16, 128, 0, stream>>>(
                xin, nullptr, Wself, nullptr, bself, nullptr,
                bnP[layer][0], bnP[layer][1], bnP[layer][2], bnP[layer][3],
                CONV, nullptr, rows, K, cout, 1);
        }
        // deg 1..10 blocks: dual-matrix (self + rel) GEMM
        for (int d = 1; d <= MAX_DEG; ++d) {
            int rows = H_COUNTS[d], start = H_STARTS[d];
            const u16*   Wself = wbase + (size_t)(2 * d - 1) * matStride;
            const float* bself = B + (size_t)(2 * d - 1) * cout;
            const u16*   Wrel  = wbase + (size_t)(2 * (d - 1)) * matStride;
            const float* brel  = B + (size_t)(2 * (d - 1)) * cout;
            gemm_bf16_kernel<true><<<(rows + 15) / 16, 128, 0, stream>>>(
                xin + (size_t)start * K, REL + (size_t)start * K, Wself, Wrel, bself, brel,
                bnP[layer][0], bnP[layer][1], bnP[layer][2], bnP[layer][3],
                CONV + (size_t)start * cout, nullptr, rows, K, cout, 1);
        }
        hipMemcpyAsync(POOL, CONV, (size_t)H_COUNTS[0] * cout * sizeof(u16),
                       hipMemcpyDeviceToDevice, stream);
        for (int d = 1; d <= MAX_DEG; ++d) {
            int rows = H_COUNTS[d];
            int total = rows * 32;                    // 64 feats = 32 uint pairs
            pool_kernel<<<(total + 255) / 256, 256, 0, stream>>>(
                CONV, adj[d - 1], POOL, rows, d, H_STARTS[d], 32);
        }
    }

    // -------------------- dense 64->128 + ReLU + BN2 (f32 out) --------------------
    gemm_bf16_kernel<false><<<(N_ATOMS + 15) / 16, 256, 0, stream>>>(
        POOL, nullptr, wp_dense, nullptr, dense_b, nullptr,
        bnP[2][0], bnP[2][1], bnP[2][2], bnP[2][3],
        nullptr, DENSE, N_ATOMS, 64, 128, 1);

    // -------------------- segment sum/max + tanh -> nf --------------------
    {
        int n = BATCH * 128;
        seg_init_kernel<<<(n + 255) / 256, 256, 0, stream>>>(SEGS, SEGM, n);
        int total = N_ATOMS * 128;
        segment_kernel<<<(total + 255) / 256, 256, 0, stream>>>(
            DENSE, membership, SEGS, SEGM, N_ATOMS, 128);
        int nfn = BATCH * 256;
        nf_kernel<<<(nfn + 255) / 256, 256, 0, stream>>>(SEGS, SEGM, NF, NFB, BATCH, 128);
    }

    // -------------------- FC heads --------------------
    gemm_bf16_kernel<false><<<BATCH / 16, 512, 0, stream>>>(   // 256->256, ReLU (bf16 out)
        NFB, nullptr, wp_fd0, nullptr, fd0_b, nullptr,
        nullptr, nullptr, nullptr, nullptr, HB, nullptr, BATCH, 256, 256, 1);

    gemm_bf16_kernel<false><<<BATCH / 16, 64, 0, stream>>>(    // 256->24 logits (f32 out)
        HB, nullptr, wp_rd, nullptr, rd_b, nullptr,
        nullptr, nullptr, nullptr, nullptr, nullptr, OUT_LOGITS, BATCH, 256, 24, 0);

    int npairs = BATCH * 12;
    softmax_kernel<<<(npairs + 255) / 256, 256, 0, stream>>>(OUT_LOGITS, OUT_SOFT, npairs);
}
